// TC_76501957477000
// MI455X (gfx1250) — matrix operand order repaired and checked by reading, not compiled, not run on hardware
//
#include <hip/hip_runtime.h>
#include <hip/hip_bf16.h>

typedef __attribute__((ext_vector_type(16))) _Float16 v16h;
typedef __attribute__((ext_vector_type(8)))  float    v8f;

#define N1C 131072
#define E1C (16 * N1C)
#define N2C 16384
#define E2C (16 * N2C)

// ---------------- float atomic max (sign-split trick; buffer init to -inf) ----
__device__ __forceinline__ void atomicMaxF32(float* addr, float v) {
    if (v >= 0.0f) atomicMax((int*)addr, __float_as_int(v));
    else           atomicMin((unsigned int*)addr, __float_as_uint(v));
}

// ---------------- tiny utility kernels ---------------------------------------
__global__ void k_fill(float* __restrict__ p, float v, int m) {
    int i = blockIdx.x * blockDim.x + threadIdx.x;
    if (i < m) p[i] = v;
}

__global__ void k_fix_neginf(float* __restrict__ p, int m) {
    int i = blockIdx.x * blockDim.x + threadIdx.x;
    if (i >= m) return;
    float v = p[i];
    if (!__builtin_isfinite(v)) p[i] = 0.0f;
}

__global__ void k_x_to_h16(const float* __restrict__ x, _Float16* __restrict__ hf16, int n) {
    int i = blockIdx.x * blockDim.x + threadIdx.x;  // over n*32
    if (i >= n * 32) return;
    int c = i & 31;
    hf16[i] = (_Float16)((c == 0) ? x[i >> 5] : 0.0f);
}

// Pack Wq|Wk|Wv (fp32 [ci,co]) into TRANSPOSED f16 panel Wt[ldW x 32]
// (K contiguous per output column -> B fragment loads become 2x b128).
__global__ void k_build_wcat(const float* __restrict__ Wq, const float* __restrict__ Wk,
                             const float* __restrict__ Wv, _Float16* __restrict__ Wt,
                             int ci, int co, int ldW) {
    int i = blockIdx.x * blockDim.x + threadIdx.x;
    if (i >= 32 * ldW) return;
    int c = i / 32, k = i % 32;          // Wt[c*32 + k]
    float v = 0.0f;
    if (k < ci) {
        if      (c <     co) v = Wq[k * co + c];
        else if (c < 2 * co) v = Wk[k * co + (c - co)];
        else if (c < 3 * co) v = Wv[k * co + (c - 2 * co)];
    }
    Wt[i] = (_Float16)v;
}

// ---------------- WMMA GEMM strip: Y[16 x LDW] = X16[16 x 32] @ W[32 x LDW] ---
// One wave per 16-row strip; A fragment loaded once, LDW/16 WMMAs unrolled.
// EXEC all-ones, no divergence around the WMMAs.
template <int LDW>
__global__ __launch_bounds__(32) void k_wmma_gemm_strip(const _Float16* __restrict__ X,
                                                        const _Float16* __restrict__ Wt,
                                                        float* __restrict__ Y) {
    int mt   = blockIdx.x;
    int lane = threadIdx.x & 31;
    int l15  = lane & 15;
    int kSel = lane >> 4;                 // 0: K{0..15}  1: K{16..31}
    const _Float16* rowA = X + (size_t)(mt * 16 + l15) * 32;
    v16h a;
#pragma unroll
    for (int j = 0; j < 16; ++j) a[j] = rowA[kSel * 16 + j];          // 2x b128
#pragma unroll
    for (int nt = 0; nt < LDW / 16; ++nt) {
        int ncol = nt * 16 + l15;
        const _Float16* colB = Wt + (size_t)ncol * 32 + kSel * 16;    // K contiguous
        v16h b;
#pragma unroll
        for (int j = 0; j < 16; ++j) b[j] = colB[j];                  // 2x b128
        v8f c = {};
        c = __builtin_amdgcn_wmma_f32_16x16x32_f16(false, a, false, b, (short)0, c,
                                                   false, false);
#pragma unroll
        for (int r = 0; r < 8; ++r) {
            int mo = mt * 16 + r + kSel * 8;  // D layout: lanes16-31 hold M=r+8
            Y[(size_t)mo * LDW + ncol] = c[r];
        }
    }
}

// ---------------- edge kernels ------------------------------------------------
// a_e = scale * dot(q[dst], k[src] + ea@We);  segment max via float atomic max
__global__ void k_edge_logits(const int* __restrict__ src, const int* __restrict__ dst,
                              const float* __restrict__ qkv, int ldq, int co,
                              const float* __restrict__ ea, const float* __restrict__ We,
                              float* __restrict__ a, float* __restrict__ amax,
                              float scale, int E) {
    int e = blockIdx.x * blockDim.x + threadIdx.x;
    if (e >= E) return;
    int s = src[e], d = dst[e];
    const float* q = qkv + (size_t)d * ldq;
    const float* k = qkv + (size_t)s * ldq + co;
    float a0 = ea[e * 3 + 0], a1 = ea[e * 3 + 1], a2 = ea[e * 3 + 2];
    float acc = 0.0f;
    for (int c = 0; c < co; ++c) {
        float ec = a0 * We[c] + a1 * We[co + c] + a2 * We[2 * co + c];
        acc += q[c] * (k[c] + ec);
    }
    acc *= scale;
    a[e] = acc;
    atomicMaxF32(&amax[d], acc);
}

__global__ void k_edge_exp(const int* __restrict__ dst, float* __restrict__ a,
                           const float* __restrict__ amax, float* __restrict__ den, int E) {
    int e = blockIdx.x * blockDim.x + threadIdx.x;
    if (e >= E) return;
    int d = dst[e];
    float ex = __expf(a[e] - amax[d]);
    a[e] = ex;                              // overwrite logits with exp in place
    atomicAdd(&den[d], ex);
}

// y[dst,c] += (v[src,c] + e_c) * ex/den ; one thread per (edge,channel)
__global__ void k_edge_scatter(const int* __restrict__ src, const int* __restrict__ dst,
                               const float* __restrict__ qkv, int ldq, int co,
                               const float* __restrict__ ea, const float* __restrict__ We,
                               const float* __restrict__ ex, const float* __restrict__ den,
                               float* __restrict__ y, long long total) {
    long long i = (long long)blockIdx.x * blockDim.x + threadIdx.x;
    if (i >= total) return;
    int e = (int)(i / co), c = (int)(i % co);
    int s = src[e], d = dst[e];
    float w  = ex[e] / fmaxf(den[d], 1e-16f);
    float ec = ea[e * 3 + 0] * We[c] + ea[e * 3 + 1] * We[co + c] + ea[e * 3 + 2] * We[2 * co + c];
    float vv = qkv[(size_t)s * ldq + 2 * co + c] + ec;
    atomicAdd(&y[(size_t)d * 32 + c], vv * w);
}

// ---------------- ELU + per-channel BN stats (block partial -> atomics) -------
__global__ __launch_bounds__(256) void k_elu_stats(float* __restrict__ y,
                                                   float* __restrict__ stats, int n) {
    int c = threadIdx.x & 31;
    int g = threadIdx.x >> 5;          // 8 row-groups per block
    float s0 = 0.0f, s1 = 0.0f;
    for (int r = blockIdx.x * 8 + g; r < n; r += gridDim.x * 8) {
        float v = y[(size_t)r * 32 + c];
        v = v > 0.0f ? v : expm1f(v);
        y[(size_t)r * 32 + c] = v;
        s0 += v; s1 += v * v;
    }
    __shared__ float ls0[8][32], ls1[8][32];
    ls0[g][c] = s0; ls1[g][c] = s1;
    __syncthreads();
    if (g == 0) {
        float t0 = 0.0f, t1 = 0.0f;
        for (int i = 0; i < 8; ++i) { t0 += ls0[i][c]; t1 += ls1[i][c]; }
        atomicAdd(&stats[c], t0);
        atomicAdd(&stats[32 + c], t1);
    }
}

// BN apply (+optional residual, +optional skip save, +f16 for next WMMA GEMM)
__global__ void k_bn_apply(const float* __restrict__ y, const float* __restrict__ stats,
                           const float* __restrict__ g, const float* __restrict__ b,
                           const float* __restrict__ resid, float* __restrict__ sc_out,
                           float* __restrict__ h_out, _Float16* __restrict__ hf16_out,
                           int n, int co) {
    int i = blockIdx.x * blockDim.x + threadIdx.x;  // n*32
    if (i >= n * 32) return;
    int c = i & 31;
    float hv = 0.0f;
    if (c < co) {
        float mean = stats[c] / (float)n;
        float var  = stats[32 + c] / (float)n - mean * mean;
        hv = g[c] * (y[i] - mean) * rsqrtf(var + 1e-5f) + b[c];
        if (resid) hv += resid[i];
    }
    if (sc_out) sc_out[i] = hv;
    h_out[i] = hv;
    hf16_out[i] = (_Float16)hv;
}

// ---------------- pooling -----------------------------------------------------
__global__ void k_pool_max(const float* __restrict__ h, const int* __restrict__ cluster,
                           float* __restrict__ h2, int n) {
    int i = blockIdx.x * blockDim.x + threadIdx.x;  // n*32
    if (i >= n * 32) return;
    int node = i >> 5, c = i & 31;
    atomicMaxF32(&h2[(size_t)cluster[node] * 32 + c], h[i]);
}

__global__ void k_pool_pos(const float* __restrict__ pos, const int* __restrict__ cluster,
                           float* __restrict__ possum, float* __restrict__ cnt, int n) {
    int i = blockIdx.x * blockDim.x + threadIdx.x;
    if (i >= n) return;
    int cl = cluster[i];
    atomicAdd(&cnt[cl], 1.0f);
    atomicAdd(&possum[cl * 3 + 0], pos[i * 3 + 0]);
    atomicAdd(&possum[cl * 3 + 1], pos[i * 3 + 1]);
    atomicAdd(&possum[cl * 3 + 2], pos[i * 3 + 2]);
}

__global__ void k_pos_div(float* __restrict__ possum, const float* __restrict__ cnt, int m) {
    int i = blockIdx.x * blockDim.x + threadIdx.x;  // n2*3
    if (i >= m) return;
    possum[i] /= fmaxf(cnt[i / 3], 1.0f);
}

__global__ void k_cart(const int* __restrict__ src, const int* __restrict__ dst,
                       const float* __restrict__ pos2, float* __restrict__ ea2,
                       unsigned int* __restrict__ cmax, int E) {
    int i = blockIdx.x * blockDim.x + threadIdx.x;  // E*3
    if (i >= E * 3) return;
    int e = i / 3, d = i % 3;
    float v = pos2[src[e] * 3 + d] - pos2[dst[e] * 3 + d];
    ea2[i] = v;
    atomicMax(cmax, __float_as_uint(fabsf(v)));     // non-negative -> uint-ordered
}

__global__ void k_cart_norm(float* __restrict__ ea2, const unsigned int* __restrict__ cmax, int m) {
    int i = blockIdx.x * blockDim.x + threadIdx.x;
    if (i >= m) return;
    float M = __uint_as_float(*cmax);
    ea2[i] = ea2[i] / (2.0f * M) + 0.5f;
}

__global__ void k_fix_prep(float* __restrict__ h2, float* __restrict__ sc2,
                           _Float16* __restrict__ h2f16, int m) {
    int i = blockIdx.x * blockDim.x + threadIdx.x;
    if (i >= m) return;
    float v = h2[i];
    if (!__builtin_isfinite(v)) v = 0.0f;
    h2[i] = v; sc2[i] = v; h2f16[i] = (_Float16)v;
}

// out[b,o] = sum_{k,c} xp[b*16+k, c] * fc[(k*32+c)*2 + o]
__global__ void k_fc(const float* __restrict__ xp, const float* __restrict__ fc,
                     float* __restrict__ out) {
    int t = threadIdx.x;
    if (t >= 32) return;
    int bb = t >> 1, o = t & 1;
    float acc = 0.0f;
    for (int k = 0; k < 16; ++k)
        for (int c = 0; c < 32; ++c)
            acc += xp[(bb * 16 + k) * 32 + c] * fc[(k * 32 + c) * 2 + o];
    out[bb * 2 + o] = acc;
}

// ---------------- one TransformerConv + ELU + BN ------------------------------
static void run_conv(int n, int E, int ci, int co,
                     const _Float16* xf16,
                     const float* Wq, const float* Wk, const float* Wv,
                     const float* We, const float* g, const float* b,
                     const int* src, const int* dst, const float* ea,
                     _Float16* wcat, float* qkv, float* aex, float* amax,
                     float* den, float* stats, float* y,
                     const float* resid, float* sc_out,
                     float* h_out, _Float16* hf16_out, hipStream_t stream) {
    int ldW = ((3 * co + 15) / 16) * 16;
    int wn  = 32 * ldW;
    k_build_wcat<<<(wn + 255) / 256, 256, 0, stream>>>(Wq, Wk, Wv, wcat, ci, co, ldW);

    int gblocks = n / 16;
    if (ldW == 32)
        k_wmma_gemm_strip<32><<<gblocks, 32, 0, stream>>>(xf16, wcat, qkv);
    else if (ldW == 48)
        k_wmma_gemm_strip<48><<<gblocks, 32, 0, stream>>>(xf16, wcat, qkv);
    else
        k_wmma_gemm_strip<96><<<gblocks, 32, 0, stream>>>(xf16, wcat, qkv);

    k_fill<<<(n + 255) / 256, 256, 0, stream>>>(amax, -__builtin_huge_valf(), n);
    hipMemsetAsync(den, 0, (size_t)n * sizeof(float), stream);
    hipMemsetAsync(y, 0, (size_t)n * 32 * sizeof(float), stream);
    hipMemsetAsync(stats, 0, 64 * sizeof(float), stream);

    float scale = 1.0f / sqrtf((float)co);
    k_edge_logits<<<(E + 255) / 256, 256, 0, stream>>>(src, dst, qkv, ldW, co, ea, We,
                                                       aex, amax, scale, E);
    k_fix_neginf<<<(n + 255) / 256, 256, 0, stream>>>(amax, n);
    k_edge_exp<<<(E + 255) / 256, 256, 0, stream>>>(dst, aex, amax, den, E);

    long long total = (long long)E * co;
    k_edge_scatter<<<(int)((total + 255) / 256), 256, 0, stream>>>(src, dst, qkv, ldW, co,
                                                                   ea, We, aex, den, y, total);
    int sblocks = (n + 7) / 8; if (sblocks > 4096) sblocks = 4096;
    k_elu_stats<<<sblocks, 256, 0, stream>>>(y, stats, n);
    k_bn_apply<<<((n * 32) + 255) / 256, 256, 0, stream>>>(y, stats, g, b, resid, sc_out,
                                                           h_out, hf16_out, n, co);
}

// ---------------- entry -------------------------------------------------------
extern "C" void kernel_launch(void* const* d_in, const int* in_sizes, int n_in,
                              void* d_out, int out_size, void* d_ws, size_t ws_size,
                              hipStream_t stream) {
    (void)in_sizes; (void)n_in; (void)out_size; (void)ws_size;
    const int N1 = N1C, E1 = E1C, N2 = N2C, E2 = E2C;

    const float* x   = (const float*)d_in[0];
    const float* pos = (const float*)d_in[1];
    const float* ea1 = (const float*)d_in[2];
    auto P = [&](int l, int j) { return (const float*)d_in[3 + 6 * l + j]; };
    const float* fc  = (const float*)d_in[45];
    const int*   ei1 = (const int*)d_in[46];
    const int*   cl5 = (const int*)d_in[47];
    const int*   ei2 = (const int*)d_in[48];
    const int*   cl7 = (const int*)d_in[49];
    float* out = (float*)d_out;

    // bump allocator over workspace
    char* w = (char*)d_ws;
    auto alloc = [&](size_t bytes) -> void* {
        void* p = (void*)w;
        w += (bytes + 255) & ~(size_t)255;
        return p;
    };
    float*    h      = (float*)alloc((size_t)N1 * 32 * 4);
    float*    y      = (float*)alloc((size_t)N1 * 32 * 4);
    float*    sc     = (float*)alloc((size_t)N1 * 32 * 4);
    float*    qkv    = (float*)alloc((size_t)N1 * 96 * 4);
    _Float16* hf16   = (_Float16*)alloc((size_t)N1 * 32 * 2);
    float*    aex    = (float*)alloc((size_t)E1 * 4);
    float*    amax   = (float*)alloc((size_t)N1 * 4);
    float*    den    = (float*)alloc((size_t)N1 * 4);
    float*    stats  = (float*)alloc(64 * 4);
    _Float16* wcat   = (_Float16*)alloc(32 * 96 * 2);
    float*    h2     = (float*)alloc((size_t)N2 * 32 * 4);
    float*    sc2    = (float*)alloc((size_t)N2 * 32 * 4);
    _Float16* h2f16  = (_Float16*)alloc((size_t)N2 * 32 * 2);
    float*    possum = (float*)alloc((size_t)N2 * 3 * 4);
    float*    cnt    = (float*)alloc((size_t)N2 * 4);
    float*    ea2    = (float*)alloc((size_t)E2 * 3 * 4);
    unsigned* cmax   = (unsigned*)alloc(4);
    float*    xp     = (float*)alloc(256 * 32 * 4);

    const int* s1 = ei1, * d1 = ei1 + E1;
    const int* s2 = ei2, * d2 = ei2 + E2;

    // feature ladder on the fine graph
    k_x_to_h16<<<((N1 * 32) + 255) / 256, 256, 0, stream>>>(x, hf16, N1);
    // conv1: 1 -> 8
    run_conv(N1, E1, 1, 8,  hf16, P(0,0), P(0,1), P(0,2), P(0,3), P(0,4), P(0,5),
             s1, d1, ea1, wcat, qkv, aex, amax, den, stats, y,
             nullptr, nullptr, h, hf16, stream);
    // conv2: 8 -> 16 ; save skip
    run_conv(N1, E1, 8, 16, hf16, P(1,0), P(1,1), P(1,2), P(1,3), P(1,4), P(1,5),
             s1, d1, ea1, wcat, qkv, aex, amax, den, stats, y,
             nullptr, sc, h, hf16, stream);
    // conv3: 16 -> 16
    run_conv(N1, E1, 16, 16, hf16, P(2,0), P(2,1), P(2,2), P(2,3), P(2,4), P(2,5),
             s1, d1, ea1, wcat, qkv, aex, amax, den, stats, y,
             nullptr, nullptr, h, hf16, stream);
    // conv4: 16 -> 16 ; add skip after BN
    run_conv(N1, E1, 16, 16, hf16, P(3,0), P(3,1), P(3,2), P(3,3), P(3,4), P(3,5),
             s1, d1, ea1, wcat, qkv, aex, amax, den, stats, y,
             sc, nullptr, h, hf16, stream);
    // conv5: 16 -> 32
    run_conv(N1, E1, 16, 32, hf16, P(4,0), P(4,1), P(4,2), P(4,3), P(4,4), P(4,5),
             s1, d1, ea1, wcat, qkv, aex, amax, den, stats, y,
             nullptr, nullptr, h, hf16, stream);

    // voxel max-pool + mean positions + normalized Cartesian edge attrs
    k_fill<<<((N2 * 32) + 255) / 256, 256, 0, stream>>>(h2, -__builtin_huge_valf(), N2 * 32);
    hipMemsetAsync(possum, 0, (size_t)N2 * 3 * 4, stream);
    hipMemsetAsync(cnt, 0, (size_t)N2 * 4, stream);
    hipMemsetAsync(cmax, 0, 4, stream);
    k_pool_max<<<((N1 * 32) + 255) / 256, 256, 0, stream>>>(h, cl5, h2, N1);
    k_pool_pos<<<(N1 + 255) / 256, 256, 0, stream>>>(pos, cl5, possum, cnt, N1);
    k_pos_div<<<((N2 * 3) + 255) / 256, 256, 0, stream>>>(possum, cnt, N2 * 3);
    k_cart<<<((E2 * 3) + 255) / 256, 256, 0, stream>>>(s2, d2, possum, ea2, cmax, E2);
    k_cart_norm<<<((E2 * 3) + 255) / 256, 256, 0, stream>>>(ea2, cmax, E2 * 3);
    k_fix_prep<<<((N2 * 32) + 255) / 256, 256, 0, stream>>>(h2, sc2, h2f16, N2 * 32);

    // coarse graph: conv6, conv7 (+skip after BN7)
    run_conv(N2, E2, 32, 32, h2f16, P(5,0), P(5,1), P(5,2), P(5,3), P(5,4), P(5,5),
             s2, d2, ea2, wcat, qkv, aex, amax, den, stats, y,
             nullptr, nullptr, h2, h2f16, stream);
    run_conv(N2, E2, 32, 32, h2f16, P(6,0), P(6,1), P(6,2), P(6,3), P(6,4), P(6,5),
             s2, d2, ea2, wcat, qkv, aex, amax, den, stats, y,
             sc2, nullptr, h2, h2f16, stream);

    // final max-pool to [B*K,32] and FC to [B,2]
    k_fill<<<((256 * 32) + 255) / 256, 256, 0, stream>>>(xp, -__builtin_huge_valf(), 256 * 32);
    k_pool_max<<<((N2 * 32) + 255) / 256, 256, 0, stream>>>(h2, cl7, xp, N2);
    k_fix_neginf<<<((256 * 32) + 255) / 256, 256, 0, stream>>>(xp, 256 * 32);
    k_fc<<<1, 32, 0, stream>>>(xp, fc, out);
}
